// SCT_Memory_25494925869416
// MI455X (gfx1250) — matrix operand order repaired and checked by reading, not compiled, not run on hardware
//
#include <hip/hip_runtime.h>
#include <hip/hip_bf16.h>

// ---------------- problem constants ----------------
#define TEMP_F     0.07f
#define INV_TEMP   (1.0f / 0.07f)
#define MOM        0.2f
#define NCAM       8
#define NB         256      // batch
#define NC         65536    // classes
#define ND         2048     // feature dim

// ---------------- GEMM tiling ----------------
#define BM 128
#define BN 128
#define BK 32
#define LDP 40               // LDS row pitch in bf16 elements (80B: 16B aligned, conflict-free)

typedef __attribute__((ext_vector_type(16))) __bf16 v16bf;
typedef __attribute__((ext_vector_type(8)))  float  v8f;
typedef __attribute__((ext_vector_type(4)))  float  v4f;   // native vector for NT stores

struct U32x8 { uint4 a, b; };

static __device__ __forceinline__ unsigned short f32_to_bf16_rne(float x) {
    unsigned int u = __builtin_bit_cast(unsigned int, x);
    unsigned int r = u + 0x7FFFu + ((u >> 16) & 1u);
    return (unsigned short)(r >> 16);
}
static __device__ __forceinline__ unsigned int pack2(unsigned short a, unsigned short b) {
    return (unsigned int)a | ((unsigned int)b << 16);
}
// packed fp32x2 -> bf16x2 (RNE); prefers the HW packed convert
static __device__ __forceinline__ unsigned int cvt_pk_bf16(float a, float b) {
#if __has_builtin(__builtin_amdgcn_cvt_pk_bf16_f32)
    return __builtin_bit_cast(unsigned int, __builtin_amdgcn_cvt_pk_bf16_f32(a, b));
#else
    return pack2(f32_to_bf16_rne(a), f32_to_bf16_rne(b));
#endif
}
static __device__ __forceinline__ v16bf make_frag(uint4 a, uint4 b) {
    U32x8 u; u.a = a; u.b = b;
    return __builtin_bit_cast(v16bf, u);
}

// split 16 fp32 (four float4) into bf16 hi + bf16 residual-lo, 32B each to LDS
static __device__ __forceinline__ void split_store16(unsigned short* hiP,
                                                     unsigned short* loP,
                                                     const float4* v) {
    unsigned int hw[8], lw[8];
#pragma unroll
    for (int q = 0; q < 4; ++q) {
        const float c[4] = { v[q].x, v[q].y, v[q].z, v[q].w };
#pragma unroll
        for (int p = 0; p < 2; ++p) {
            float a = c[2 * p], b = c[2 * p + 1];
            unsigned int ph = cvt_pk_bf16(a, b);
            float ra = a - __builtin_bit_cast(float, ph << 16);
            float rb = b - __builtin_bit_cast(float, ph & 0xFFFF0000u);
            hw[2 * q + p] = ph;
            lw[2 * q + p] = cvt_pk_bf16(ra, rb);
        }
    }
    uint4 u;
    u.x = hw[0]; u.y = hw[1]; u.z = hw[2]; u.w = hw[3]; *(uint4*)(hiP)     = u;
    u.x = hw[4]; u.y = hw[5]; u.z = hw[6]; u.w = hw[7]; *(uint4*)(hiP + 8) = u;
    u.x = lw[0]; u.y = lw[1]; u.z = lw[2]; u.w = lw[3]; *(uint4*)(loP)     = u;
    u.x = lw[4]; u.y = lw[5]; u.z = lw[6]; u.w = lw[7]; *(uint4*)(loP + 8) = u;
}

// ============================================================================
// Fused GEMM (bf16x3 split on v_wmma_f32_16x16x32_bf16) + camera-masked
// exp-sum + write-through copy of class_memory -> new_memory (NT stores).
// grid: (NB/BM, NC/BN)  [m-tiles fastest so both consumers of a B tile are
// adjacent in dispatch order -> B streamed from HBM once], block: 256 (8 waves)
// ============================================================================
__global__ __launch_bounds__(256)
void sct_gemm_expsum(const float* __restrict__ F,      // [NB][ND]
                     const float* __restrict__ M,      // [NC][ND]
                     const int*   __restrict__ cams,   // [NB]
                     const int*   __restrict__ ccam,   // [NC]
                     float*       __restrict__ row_sums, // [NB]
                     float*       __restrict__ out_mem)  // [NC][ND]
{
    __shared__ __align__(16) unsigned short Ahi[BM * LDP];
    __shared__ __align__(16) unsigned short Alo[BM * LDP];
    __shared__ __align__(16) unsigned short Bhi[BN * LDP];
    __shared__ __align__(16) unsigned short Blo[BN * LDP];

    const int tid  = threadIdx.x;
    const int wave = tid >> 5;
    const int lane = tid & 31;
    const int half = lane >> 4;
    const int l16  = lane & 15;

    const int mblock = blockIdx.x * BM;
    const int nblock = blockIdx.y * BN;
    const bool do_copy = (blockIdx.x == 0);

    v8f acc[8];
#pragma unroll
    for (int t = 0; t < 8; ++t)
#pragma unroll
        for (int v = 0; v < 8; ++v) acc[t][v] = 0.0f;

    // cooperative-load coordinates: 2 threads per row, 16 fp32 each
    const int lr = tid >> 1;          // 0..127
    const int lc = (tid & 1) * 16;    // 0 or 16

    const float* __restrict__ pa_row = F + (size_t)(mblock + lr) * ND + lc;
    const float* __restrict__ pb_row = M + (size_t)(nblock + lr) * ND + lc;
    float* __restrict__ pc_row = out_mem + (size_t)(nblock + lr) * ND + lc;

    // ---- software pipeline: prefetch k-step 0 into registers ----
    float4 fav[4], fbv[4];
#pragma unroll
    for (int q = 0; q < 4; ++q) {
        fav[q] = ((const float4*)pa_row)[q];
        fbv[q] = ((const float4*)pb_row)[q];
    }

    for (int k0 = 0; k0 < ND; k0 += BK) {
        // ---- stage: split fp32 -> bf16 hi/lo in LDS (converted once per WG) ----
        split_store16(&Ahi[lr * LDP + lc], &Alo[lr * LDP + lc], fav);
        split_store16(&Bhi[lr * LDP + lc], &Blo[lr * LDP + lc], fbv);
        if (do_copy) {
            v4f* dst = (v4f*)(pc_row + k0);
#pragma unroll
            for (int q = 0; q < 4; ++q)
                __builtin_nontemporal_store(__builtin_bit_cast(v4f, fbv[q]),
                                            &dst[q]);  // write-once stream
        }
        __syncthreads();

        // ---- prefetch next k-step while WMMAs run ----
        if (k0 + BK < ND) {
#pragma unroll
            for (int q = 0; q < 4; ++q) {
                fav[q] = ((const float4*)(pa_row + k0 + BK))[q];
                fbv[q] = ((const float4*)(pb_row + k0 + BK))[q];
            }
        }

        // ---- A fragment (ISA 16-bit A 16x32 layout) ----
        const int aoff = (wave * 16 + l16) * LDP;
        v16bf afh = make_frag(*(const uint4*)&Ahi[aoff + 8 * half],
                              *(const uint4*)&Ahi[aoff + 16 + 8 * half]);
        v16bf afl = make_frag(*(const uint4*)&Alo[aoff + 8 * half],
                              *(const uint4*)&Alo[aoff + 16 + 8 * half]);

#pragma unroll
        for (int t = 0; t < 8; ++t) {
            // ---- B fragment (ISA 16-bit B 32x16 layout): k = 8*half+s, 16+8*half+s ----
            const int boff = (t * 16 + l16) * LDP + 8 * half;
            v16bf bfh = make_frag(*(const uint4*)&Bhi[boff],
                                  *(const uint4*)&Bhi[boff + 16]);
            v16bf bfl = make_frag(*(const uint4*)&Blo[boff],
                                  *(const uint4*)&Blo[boff + 16]);
            acc[t] = __builtin_amdgcn_wmma_f32_16x16x32_bf16(
                false, afh, false, bfh, (short)0, acc[t], false, false);
            acc[t] = __builtin_amdgcn_wmma_f32_16x16x32_bf16(
                false, afh, false, bfl, (short)0, acc[t], false, false);
            acc[t] = __builtin_amdgcn_wmma_f32_16x16x32_bf16(
                false, afl, false, bfh, (short)0, acc[t], false, false);
        }
        __syncthreads();
    }

    // ---- epilogue: camera-masked exp-sum, reduced per row ----
    // C/D layout: lane l holds col (l&15); VGPR v holds row v + 8*(l>>4)
    const int mbase = mblock + wave * 16 + 8 * half;
    float rsum[8];
    int camr[8];
#pragma unroll
    for (int v = 0; v < 8; ++v) { rsum[v] = 0.0f; camr[v] = cams[mbase + v]; }

#pragma unroll
    for (int t = 0; t < 8; ++t) {
        const int col = nblock + t * 16 + l16;
        const int cc  = ccam[col];
#pragma unroll
        for (int v = 0; v < 8; ++v) {
            float e = __expf(acc[t][v] * INV_TEMP);
            rsum[v] += (cc == camr[v]) ? e : 0.0f;
        }
    }
#pragma unroll
    for (int v = 0; v < 8; ++v) {
        float s = rsum[v];
        s += __shfl_xor(s, 1, 32);
        s += __shfl_xor(s, 2, 32);
        s += __shfl_xor(s, 4, 32);
        s += __shfl_xor(s, 8, 32);           // sums the 16 lanes of each half
        if (l16 == 0) atomicAdd(&row_sums[mbase + v], s);
    }
}

// ============================================================================
// pos[i] = dot(F[i], M[labels[i]]) / TEMP     (256 blocks x 256 threads)
// ============================================================================
__global__ __launch_bounds__(256)
void sct_pos(const float* __restrict__ F, const float* __restrict__ M,
             const int* __restrict__ labels, float* __restrict__ pos)
{
    const int i = blockIdx.x, t = threadIdx.x;
    const int y = labels[i];
    const float* f = F + (size_t)i * ND;
    const float* m = M + (size_t)y * ND;
    float s = 0.0f;
    for (int j = t; j < ND; j += 256) s += f[j] * m[j];
    s += __shfl_xor(s, 1, 32);  s += __shfl_xor(s, 2, 32);
    s += __shfl_xor(s, 4, 32);  s += __shfl_xor(s, 8, 32);
    s += __shfl_xor(s, 16, 32);
    __shared__ float red[8];
    if ((t & 31) == 0) red[t >> 5] = s;
    __syncthreads();
    if (t == 0) {
        float tot = 0.0f;
#pragma unroll
        for (int w = 0; w < 8; ++w) tot += red[w];
        pos[i] = tot * INV_TEMP;
    }
}

// ============================================================================
__global__ __launch_bounds__(256)
void sct_zero(float* __restrict__ row_sums)
{
    row_sums[threadIdx.x] = 0.0f;
}

// ============================================================================
// loss = sum_i -(pos[i] - log(row_sums[i])) / count[cams[i]]
// ============================================================================
__global__ __launch_bounds__(256)
void sct_loss(const int* __restrict__ cams, const float* __restrict__ pos,
              const float* __restrict__ row_sums, float* __restrict__ out)
{
    __shared__ int cnt[NCAM];
    __shared__ float red[8];
    const int t = threadIdx.x;
    if (t < NCAM) cnt[t] = 0;
    __syncthreads();
    const int c = cams[t];
    atomicAdd(&cnt[c], 1);
    __syncthreads();
    float contrib = -(pos[t] - logf(row_sums[t])) / (float)cnt[c];
    contrib += __shfl_xor(contrib, 1, 32);
    contrib += __shfl_xor(contrib, 2, 32);
    contrib += __shfl_xor(contrib, 4, 32);
    contrib += __shfl_xor(contrib, 8, 32);
    contrib += __shfl_xor(contrib, 16, 32);
    if ((t & 31) == 0) red[t >> 5] = contrib;
    __syncthreads();
    if (t == 0) {
        float tot = 0.0f;
#pragma unroll
        for (int w = 0; w < 8; ++w) tot += red[w];
        out[0] = tot;
    }
}

// ============================================================================
// Sequential EMA chains: block s owns label labels[s] iff s is its first
// occurrence; applies all duplicates in batch order, L2-renorm each step.
// ============================================================================
__global__ __launch_bounds__(256)
void sct_ema(const float* __restrict__ F, const float* __restrict__ Mem,
             const int* __restrict__ labels, float* __restrict__ out_mem)
{
    const int s = blockIdx.x, t = threadIdx.x;
    __shared__ int sy, sfirst;
    if (t == 0) {
        const int y = labels[s];
        int first = 1;
        for (int q = 0; q < s; ++q)
            if (labels[q] == y) { first = 0; break; }
        sy = y; sfirst = first;
    }
    __syncthreads();
    if (!sfirst) return;
    const int y = sy;

    float r[8];
#pragma unroll
    for (int j = 0; j < 8; ++j) r[j] = Mem[(size_t)y * ND + j * 256 + t];

    __shared__ float red[8];
    __shared__ float binv;
    for (int q = s; q < NB; ++q) {
        if (labels[q] != y) continue;              // uniform across block
#pragma unroll
        for (int j = 0; j < 8; ++j)
            r[j] = MOM * r[j] + (1.0f - MOM) * F[(size_t)q * ND + j * 256 + t];
        float ss = 0.0f;
#pragma unroll
        for (int j = 0; j < 8; ++j) ss += r[j] * r[j];
        ss += __shfl_xor(ss, 1, 32);  ss += __shfl_xor(ss, 2, 32);
        ss += __shfl_xor(ss, 4, 32);  ss += __shfl_xor(ss, 8, 32);
        ss += __shfl_xor(ss, 16, 32);
        if ((t & 31) == 0) red[t >> 5] = ss;
        __syncthreads();
        if (t == 0) {
            float tot = 0.0f;
#pragma unroll
            for (int w = 0; w < 8; ++w) tot += red[w];
            binv = 1.0f / sqrtf(tot);
        }
        __syncthreads();
        const float inv = binv;
#pragma unroll
        for (int j = 0; j < 8; ++j) r[j] *= inv;
        __syncthreads();
    }
#pragma unroll
    for (int j = 0; j < 8; ++j) out_mem[(size_t)y * ND + j * 256 + t] = r[j];
}

// ============================================================================
extern "C" void kernel_launch(void* const* d_in, const int* in_sizes, int n_in,
                              void* d_out, int out_size, void* d_ws, size_t ws_size,
                              hipStream_t stream) {
    const float* F      = (const float*)d_in[0];   // features [256,2048]
    const int*   labels = (const int*)  d_in[1];   // [256]
    const int*   cams   = (const int*)  d_in[2];   // [256]
    const float* Mem    = (const float*)d_in[3];   // class_memory [65536,2048]
    const int*   ccam   = (const int*)  d_in[4];   // class_camera [65536]

    float* out      = (float*)d_out;
    float* loss_out = out;          // scalar loss first
    float* out_mem  = out + 1;      // then new_memory [65536,2048]

    float* row_sums = (float*)d_ws;         // [256]
    float* pos      = row_sums + NB;        // [256]

    sct_zero<<<1, 256, 0, stream>>>(row_sums);
    sct_pos<<<NB, 256, 0, stream>>>(F, Mem, labels, pos);
    // GEMM also performs the bulk class_memory -> new_memory copy (NT stores)
    sct_gemm_expsum<<<dim3(NB / BM, NC / BN), 256, 0, stream>>>(
        F, Mem, cams, ccam, row_sums, out_mem);
    sct_loss<<<1, 256, 0, stream>>>(cams, pos, row_sums, loss_out);
    sct_ema<<<NB, 256, 0, stream>>>(F, Mem, labels, out_mem);
}
